// ULW_prd_Net_46840913330482
// MI455X (gfx1250) — matrix operand order repaired and checked by reading, not compiled, not run on hardware
//
#include <hip/hip_runtime.h>
#include <hip/hip_bf16.h>

typedef __attribute__((ext_vector_type(16))) _Float16 v16h;
typedef __attribute__((ext_vector_type(8)))  _Float16 v8h;
typedef __attribute__((ext_vector_type(8)))  float    v8f;
typedef __attribute__((ext_vector_type(4)))  unsigned int u32x4;
typedef __attribute__((ext_vector_type(8)))  int i32x8;
typedef __attribute__((ext_vector_type(4)))  int i32x4;

#if __has_builtin(__builtin_amdgcn_tensor_load_to_lds) && \
    __has_builtin(__builtin_amdgcn_s_wait_tensorcnt)
#define GEMM_TDM 1
#endif

static __device__ __forceinline__ v16h cat16(v8h lo, v8h hi) {
  return __builtin_shufflevector(lo, hi, 0, 1, 2, 3, 4, 5, 6, 7,
                                 8, 9, 10, 11, 12, 13, 14, 15);
}
static __device__ __forceinline__ v8f wmma32(v16h a, v16h b, v8f c) {
  return __builtin_amdgcn_wmma_f32_16x16x32_f16(false, a, false, b, (short)0,
                                                c, false, false);
}
// LDS B-panel layout: streamed half index -> padded half offset
// (pad 4 halves after every 512 halves == 2 DWORDs per 256 DWORDs, matching
//  the TDM descriptor's pad_interval=7 / pad_amount=1)
static __device__ __forceinline__ unsigned lds_halfoff(unsigned idx) {
  return idx + ((idx >> 9) << 2);
}

// ---------------------------------------------------------------------------
__global__ void cvt_f32_to_f16(const float* __restrict__ src,
                               _Float16* __restrict__ dst, int n) {
  int i = blockIdx.x * blockDim.x + threadIdx.x;
  int stride = gridDim.x * blockDim.x;
  for (; i < n; i += stride) dst[i] = (_Float16)src[i];
}

// transpose + convert: src f32 (K x N) -> dst f16 (Nout x Kout), zero-pad.
__global__ void transpose_cvt(const float* __restrict__ src,
                              _Float16* __restrict__ dst,
                              int K, int N, int Kout, int Nout) {
  int i = blockIdx.x * blockDim.x + threadIdx.x;
  int total = Kout * Nout;
  int stride = gridDim.x * blockDim.x;
  for (; i < total; i += stride) {
    int n = i / Kout, k = i % Kout;
    float v = (k < K && n < N) ? src[(size_t)k * N + n] : 0.0f;
    dst[i] = (_Float16)v;
  }
}

// bias12[n] = b2[n] + sum_k b1[k]*W2[k][n]
__global__ void bias12_kernel(const float* __restrict__ b1,
                              const float* __restrict__ W2,
                              const float* __restrict__ b2,
                              float* __restrict__ bias12, int D1, int N) {
  int n = blockIdx.x * blockDim.x + threadIdx.x;
  if (n >= N) return;
  float s = b2[n];
  for (int k = 0; k < D1; ++k) s += b1[k] * W2[(size_t)k * N + n];
  bias12[n] = s;
}

// ---------------------------------------------------------------------------
// WMMA GEMM:  C(MxN,f32) = A(MxK f16 row-major) x B (+ bias[n]).
// Bt is B transposed (N rows x K cols, f16). Block = 4 waves; block computes
// 128x64; each wave a 32x64 strip. The 64-row B panel is staged into LDS by
// the Tensor Data Mover (TDM) with bank-conflict padding; A is double-buffered
// from global. K mult of 32, M mult of 128, N mult of 64.
__global__ void __launch_bounds__(128)
wmma_gemm_f16(const _Float16* __restrict__ A, const _Float16* __restrict__ Bt,
              float* __restrict__ C, const float* __restrict__ bias,
              int M, int N, int K) {
  extern __shared__ _Float16 sB[];
  const int lane  = threadIdx.x & 31;
  const int wave  = threadIdx.x >> 5;
  const int mtile = blockIdx.x * 128 + wave * 32;
  const int ntile = blockIdx.y * 64;

  // ---- stage B panel (64 rows x K halves) into LDS ----
#ifdef GEMM_TDM
  if (wave == 0) {
    unsigned long long gaddr =
        (unsigned long long)(size_t)(Bt + (size_t)ntile * K);
    unsigned ldsaddr = (unsigned)(size_t)(void*)sB;  // low 32 bits = LDS offset
    u32x4 g0;
    g0[0] = 1u;                                   // count=1, user descriptor
    g0[1] = ldsaddr;                              // lds_addr (bytes)
    g0[2] = (unsigned)(gaddr & 0xffffffffu);      // global_addr[31:0]
    g0[3] = ((unsigned)(gaddr >> 32) & 0x01ffffffu) | (2u << 30);  // type=2
    i32x8 g1;
    // data_size=1(2B), pad_enable=1, pad_interval=7(256 DW), pad_amount=1(2 DW)
    g1[0] = (int)((1u << 16) | (1u << 20) | (7u << 22) | (1u << 25));
    g1[1] = (int)(((unsigned)K & 0xffffu) << 16);          // tensor_dim0 lo16
    g1[2] = (int)((((unsigned)K >> 16) & 0xffffu) | (64u << 16)); // td0 hi|td1 lo
    g1[3] = (int)(((unsigned)K & 0xffffu) << 16);          // tile_dim0 = K
    g1[4] = 64;                                            // tile_dim1 = 64
    g1[5] = K;                                             // dim0_stride lo32
    g1[6] = 0;
    g1[7] = 0;
    i32x4 gz = {0, 0, 0, 0};
#if defined(__clang_major__) && (__clang_major__ >= 23)
    i32x8 gz8 = {0, 0, 0, 0, 0, 0, 0, 0};
    __builtin_amdgcn_tensor_load_to_lds(g0, g1, gz, gz, gz8, 0);
#else
    __builtin_amdgcn_tensor_load_to_lds(g0, g1, gz, gz, 0);
#endif
    __builtin_amdgcn_s_wait_tensorcnt((short)0);
  }
#else
  {
    const _Float16* src = Bt + (size_t)ntile * K;
    int total8 = (64 * K) >> 3;
    for (int c = threadIdx.x; c < total8; c += 128) {
      unsigned idx = (unsigned)c << 3;
      *(v8h*)(sB + lds_halfoff(idx)) = *(const v8h*)(src + idx);
    }
  }
#endif
  __syncthreads();

  const int l15   = lane & 15;
  const bool hiL  = lane >= 16;
  const int koffA = hiL ? 8 : 0;   // A halves 0-7 hold K=koffA.., 8-15 hold +16
  const int koffB = hiL ? 8 : 0;   // B halves 0-7 hold K=koffB.., 8-15 hold +16

  v8f acc[2][4];
#pragma unroll
  for (int u = 0; u < 2; ++u)
#pragma unroll
    for (int t = 0; t < 4; ++t)
#pragma unroll
      for (int j = 0; j < 8; ++j) acc[u][t][j] = 0.0f;

  const _Float16* arow0 = A + (size_t)(mtile + l15) * K;
  const _Float16* arow1 = A + (size_t)(mtile + 16 + l15) * K;

  v8h a0lo = *(const v8h*)(arow0 + koffA);
  v8h a0hi = *(const v8h*)(arow0 + 16 + koffA);
  v8h a1lo = *(const v8h*)(arow1 + koffA);
  v8h a1hi = *(const v8h*)(arow1 + 16 + koffA);

  for (int k0 = 0; k0 < K; k0 += 32) {
    v16h a0 = cat16(a0lo, a0hi);
    v16h a1 = cat16(a1lo, a1hi);
    int kn = k0 + 32;
    if (kn < K) {  // prefetch next K-step while WMMAs run
      a0lo = *(const v8h*)(arow0 + kn + koffA);
      a0hi = *(const v8h*)(arow0 + kn + 16 + koffA);
      a1lo = *(const v8h*)(arow1 + kn + koffA);
      a1hi = *(const v8h*)(arow1 + kn + 16 + koffA);
    }
#pragma unroll
    for (int t = 0; t < 4; ++t) {
      unsigned idx = (unsigned)(t * 16 + l15) * (unsigned)K + k0 + koffB;
      const _Float16* bp = sB + lds_halfoff(idx);
      v8h blo = *(const v8h*)bp;
      v8h bhi = *(const v8h*)(bp + 16);
      v16h b = cat16(blo, bhi);
      acc[0][t] = wmma32(a0, b, acc[0][t]);
      acc[1][t] = wmma32(a1, b, acc[1][t]);
    }
  }

#pragma unroll
  for (int u = 0; u < 2; ++u) {
    const int crow = mtile + u * 16 + (hiL ? 8 : 0);
#pragma unroll
    for (int t = 0; t < 4; ++t) {
      int ccol = ntile + t * 16 + l15;
      float bv = bias ? bias[ccol] : 0.0f;
#pragma unroll
      for (int j = 0; j < 8; ++j)
        C[(size_t)(crow + j) * N + ccol] = acc[u][t][j] + bv;
    }
  }
}

// ---------------------------------------------------------------------------
// per-row L2 normalize, in place. One wave32 per row.
__global__ void __launch_bounds__(256)
rownorm_kernel(float* __restrict__ H, int M, int D) {
  int lane = threadIdx.x & 31;
  int wid  = (blockIdx.x * blockDim.x + threadIdx.x) >> 5;
  if (wid >= M) return;
  float* r = H + (size_t)wid * D;
  float s = 0.0f;
  for (int d = lane; d < D; d += 32) { float v = r[d]; s += v * v; }
  for (int m = 16; m > 0; m >>= 1) s += __shfl_xor(s, m, 32);
  float inv = rsqrtf(s);
  for (int d = lane; d < D; d += 32) r[d] *= inv;
}

// ---------------------------------------------------------------------------
// sequential memory-bank scan: one block per class, thread t owns dim t.
__global__ void __launch_bounds__(256)
scan_kernel(const int* __restrict__ llabel, const float* __restrict__ lfeat,
            const float* __restrict__ mbank0, const float* __restrict__ start0,
            float* __restrict__ mbank_out, int B, int D) {
  int c = blockIdx.x;
  int d = threadIdx.x;
  float row = mbank0[(size_t)c * D + d];
  bool started = (start0[c] != 0.0f);
  for (int i = 0; i < B; ++i) {
    int lab = llabel[i];
    if (lab == c) {
      float f = lfeat[(size_t)i * D + d];
      row = started ? (0.9f * row + 0.1f * f) : f;
      started = true;
    }
  }
  mbank_out[(size_t)c * D + d] = row;
}

// ---------------------------------------------------------------------------
// lscores[i] = min_c || lfeat[i] - mbank[c] ||   (wave32 per row, bank in LDS)
__global__ void __launch_bounds__(256)
lscores_kernel(const float* __restrict__ feat, const float* __restrict__ mbank,
               float* __restrict__ out, int M) {
  __shared__ float smb[40 * 256];
  for (int i = threadIdx.x; i < 40 * 256; i += blockDim.x) smb[i] = mbank[i];
  __syncthreads();
  int lane = threadIdx.x & 31;
  int wid  = (blockIdx.x * blockDim.x + threadIdx.x) >> 5;
  if (wid >= M) return;
  const float* r = feat + (size_t)wid * 256;
  float x[8];
#pragma unroll
  for (int q = 0; q < 8; ++q) x[q] = r[lane + q * 32];
  float best = 3.4e38f;
  for (int c = 0; c < 40; ++c) {
    const float* m = smb + c * 256;
    float s = 0.0f;
#pragma unroll
    for (int q = 0; q < 8; ++q) {
      float dl = x[q] - m[lane + q * 32];
      s += dl * dl;
    }
    for (int sh = 16; sh > 0; sh >>= 1) s += __shfl_xor(s, sh, 32);
    best = fminf(best, s);
  }
  if (lane == 0) out[wid] = sqrtf(best);
}

// um distances + build padded f16 row [ufeat(256) | um(40) | 0(24)] -> 320
__global__ void __launch_bounds__(256)
um_uf_kernel(const float* __restrict__ ufeat, const float* __restrict__ mbank,
             _Float16* __restrict__ ufp, int M) {
  __shared__ float smb[40 * 256];
  for (int i = threadIdx.x; i < 40 * 256; i += blockDim.x) smb[i] = mbank[i];
  __syncthreads();
  int lane = threadIdx.x & 31;
  int wid  = (blockIdx.x * blockDim.x + threadIdx.x) >> 5;
  if (wid >= M) return;
  const float* r = ufeat + (size_t)wid * 256;
  _Float16* urow = ufp + (size_t)wid * 320;
  float x[8];
#pragma unroll
  for (int q = 0; q < 8; ++q) {
    x[q] = r[lane + q * 32];
    urow[lane + q * 32] = (_Float16)x[q];
  }
  for (int k = 296 + lane; k < 320; k += 32) urow[k] = (_Float16)0.0f;
  for (int c = 0; c < 40; ++c) {
    const float* m = smb + c * 256;
    float s = 0.0f;
#pragma unroll
    for (int q = 0; q < 8; ++q) {
      float dl = x[q] - m[lane + q * 32];
      s += dl * dl;
    }
    for (int sh = 16; sh > 0; sh >>= 1) s += __shfl_xor(s, sh, 32);
    if (lane == 0) urow[256 + c] = (_Float16)sqrtf(s);
  }
}

// ---------------------------------------------------------------------------
// per-column mean + rsqrt(var+eps); stats = [mu[N] | rstd[N]]
__global__ void __launch_bounds__(256)
colstats_kernel(const float* __restrict__ Y, float* __restrict__ stats,
                int M, int N) {
  int n = blockIdx.x;
  __shared__ float ssum[256], ssq[256];
  float s = 0.0f, q = 0.0f;
  for (int i = threadIdx.x; i < M; i += blockDim.x) {
    float v = Y[(size_t)i * N + n];
    s += v; q += v * v;
  }
  ssum[threadIdx.x] = s; ssq[threadIdx.x] = q;
  __syncthreads();
  for (int st = 128; st > 0; st >>= 1) {
    if (threadIdx.x < st) {
      ssum[threadIdx.x] += ssum[threadIdx.x + st];
      ssq[threadIdx.x]  += ssq[threadIdx.x + st];
    }
    __syncthreads();
  }
  if (threadIdx.x == 0) {
    float mu  = ssum[0] / (float)M;
    float var = ssq[0] / (float)M - mu * mu;
    stats[n]     = mu;
    stats[N + n] = rsqrtf(var + 1e-5f);
  }
}

// u = relu(g*(y-mu)*rstd + be) -> f16
__global__ void bnrelu_f16_kernel(const float* __restrict__ Y,
                                  const float* __restrict__ stats,
                                  const float* __restrict__ g,
                                  const float* __restrict__ be,
                                  _Float16* __restrict__ out, int M, int N) {
  int i = blockIdx.x * blockDim.x + threadIdx.x;
  int total = M * N;
  int stride = gridDim.x * blockDim.x;
  for (; i < total; i += stride) {
    int n = i % N;
    float v = g[n] * (Y[i] - stats[n]) * stats[N + n] + be[n];
    out[i] = (_Float16)fmaxf(v, 0.0f);
  }
}

// uscores[i] = b5 + sum_k relu(bn(y2[i,k])) * W5[k]   (wave32 per row)
__global__ void __launch_bounds__(256)
final_kernel(const float* __restrict__ Y2, const float* __restrict__ stats,
             const float* __restrict__ g, const float* __restrict__ be,
             const float* __restrict__ W5, const float* __restrict__ b5,
             float* __restrict__ out, int M, int N) {
  int lane = threadIdx.x & 31;
  int wid  = (blockIdx.x * blockDim.x + threadIdx.x) >> 5;
  if (wid >= M) return;
  const float* r = Y2 + (size_t)wid * N;
  float s = 0.0f;
  for (int k = lane; k < N; k += 32) {
    float v = g[k] * (r[k] - stats[k]) * stats[N + k] + be[k];
    s += fmaxf(v, 0.0f) * W5[k];
  }
  for (int sh = 16; sh > 0; sh >>= 1) s += __shfl_xor(s, sh, 32);
  if (lane == 0) out[wid] = s + b5[0];
}

// ---------------------------------------------------------------------------
extern "C" void kernel_launch(void* const* d_in, const int* in_sizes, int n_in,
                              void* d_out, int out_size, void* d_ws,
                              size_t ws_size, hipStream_t stream) {
  const float* lufeat = (const float*)d_in[0];
  const int*   llabel = (const int*)d_in[1];
  const float* W1 = (const float*)d_in[2];
  const float* b1 = (const float*)d_in[3];
  const float* W2 = (const float*)d_in[4];
  const float* b2 = (const float*)d_in[5];
  const float* W3 = (const float*)d_in[6];
  const float* b3 = (const float*)d_in[7];
  const float* W4 = (const float*)d_in[8];
  const float* b4 = (const float*)d_in[9];
  const float* W5 = (const float*)d_in[10];
  const float* b5 = (const float*)d_in[11];
  const float* g1 = (const float*)d_in[12];
  const float* be1 = (const float*)d_in[13];
  const float* g2 = (const float*)d_in[14];
  const float* be2 = (const float*)d_in[15];
  const float* mbank0 = (const float*)d_in[16];
  const float* start0 = (const float*)d_in[17];

  constexpr int B2 = 16384, Din = 1024, D1 = 512, DIMS = 256, C = 40;
  constexpr int B = B2 / 2;          // 8192
  constexpr int K3 = 320;            // padded DIMS + C (296 -> 320)
  constexpr int H3 = 64;

  char* ws = (char*)d_ws;
  // persistent region
  _Float16* Xf16   = (_Float16*)(ws + 0);                       // 33,554,432 B
  _Float16* W1f16  = (_Float16*)(ws + 33554432);                //  1,048,576
  _Float16* W2t    = (_Float16*)(ws + 34603008);                //    262,144
  float*    W12    = (float*)   (ws + 34865152);                //  1,048,576
  _Float16* W12t   = (_Float16*)(ws + 35913728);                //    524,288
  float*    bias12 = (float*)   (ws + 36438016);                //      1,024
  float*    H      = (float*)   (ws + 36439040);                // 16,777,216 (ends 53,216,256)
  // region reusing Xf16 space after the big GEMM
  _Float16* ufp    = (_Float16*)(ws + 0);                       //  5,242,880
  _Float16* W3t    = (_Float16*)(ws + 5242880);                 //    163,840
  float*    y1     = (float*)   (ws + 5406720);                 //  8,388,608
  float*    stats1 = (float*)   (ws + 13795328);                //      2,048
  _Float16* u1f16  = (_Float16*)(ws + 13797376);                //  4,194,304
  _Float16* W4t    = (_Float16*)(ws + 17991680);                //     32,768
  float*    y2     = (float*)   (ws + 18024448);                //  2,097,152
  float*    stats2 = (float*)   (ws + 20121600);                //        512

  float* out_lscores = (float*)d_out;
  float* out_uscores = (float*)d_out + B;
  float* out_mbank   = (float*)d_out + 2 * B;

  // LDS bytes for a 64-row x K-half B panel with pad (4 halves per 512)
  auto lds_bytes = [](int K) -> size_t {
    int halves = 64 * K + ((64 * K) >> 9) * 4;
    return (size_t)halves * 2;
  };

  // 1) conversions / fused-weight prep
  cvt_f32_to_f16<<<4096, 256, 0, stream>>>(lufeat, Xf16, B2 * Din);
  cvt_f32_to_f16<<<1024, 256, 0, stream>>>(W1, W1f16, Din * D1);
  transpose_cvt<<<512, 256, 0, stream>>>(W2, W2t, D1, DIMS, D1, DIMS);
  bias12_kernel<<<1, 256, 0, stream>>>(b1, W2, b2, bias12, D1, DIMS);

  // 2) W12 = W1 @ W2 (f32 out), then transpose to f16
  {
    dim3 g(Din / 128, DIMS / 64);
    wmma_gemm_f16<<<g, 128, lds_bytes(D1), stream>>>(W1f16, W2t, W12, nullptr,
                                                     Din, DIMS, D1);
  }
  transpose_cvt<<<1024, 256, 0, stream>>>(W12, W12t, Din, DIMS, Din, DIMS);

  // 3) H = X @ W12 + bias12 ; normalize rows
  {
    dim3 g(B2 / 128, DIMS / 64);
    wmma_gemm_f16<<<g, 128, lds_bytes(Din), stream>>>(Xf16, W12t, H, bias12,
                                                      B2, DIMS, Din);
  }
  rownorm_kernel<<<B2 / 8, 256, 0, stream>>>(H, B2, DIMS);

  // 4) memory-bank scan (labeled half = rows 0..B-1)
  scan_kernel<<<C, DIMS, 0, stream>>>(llabel, H, mbank0, start0,
                                      out_mbank, B, DIMS);

  // 5) lscores
  lscores_kernel<<<B / 8, 256, 0, stream>>>(H, out_mbank, out_lscores, B);

  // 6) um + padded uf (f16), W3 transpose (reuses freed Xf16 region)
  transpose_cvt<<<320, 256, 0, stream>>>(W3, W3t, DIMS + C, 256, K3, 256);
  um_uf_kernel<<<B / 8, 256, 0, stream>>>(H + (size_t)B * DIMS, out_mbank,
                                          ufp, B);

  // 7) y1 = uf @ W3 + b3 ; BN stats ; relu -> f16
  {
    dim3 g(B / 128, 256 / 64);
    wmma_gemm_f16<<<g, 128, lds_bytes(K3), stream>>>(ufp, W3t, y1, b3,
                                                     B, 256, K3);
  }
  colstats_kernel<<<256, 256, 0, stream>>>(y1, stats1, B, 256);
  bnrelu_f16_kernel<<<2048, 256, 0, stream>>>(y1, stats1, g1, be1,
                                              u1f16, B, 256);

  // 8) y2 = u1 @ W4 + b4 ; BN stats ; fused relu/bn/dot for uscores
  transpose_cvt<<<64, 256, 0, stream>>>(W4, W4t, 256, H3, 256, H3);
  {
    dim3 g(B / 128, H3 / 64);
    wmma_gemm_f16<<<g, 128, lds_bytes(DIMS), stream>>>(u1f16, W4t, y2, b4,
                                                       B, H3, 256);
  }
  colstats_kernel<<<H3, 256, 0, stream>>>(y2, stats2, B, H3);
  final_kernel<<<B / 8, 256, 0, stream>>>(y2, stats2, g2, be2, W5, b5,
                                          out_uscores, B, H3);
}
